// Consolidation_111669149787
// MI455X (gfx1250) — matrix operand order repaired and checked by reading, not compiled, not run on hardware
//
#include <hip/hip_runtime.h>

typedef __attribute__((ext_vector_type(16))) _Float16     v16h;
typedef __attribute__((ext_vector_type(8)))  _Float16     v8h;
typedef __attribute__((ext_vector_type(8)))  float        v8f;
typedef __attribute__((ext_vector_type(4)))  unsigned int v4u;
typedef __attribute__((ext_vector_type(4)))  int          v4i;
typedef __attribute__((ext_vector_type(8)))  int          v8i;

#define T_DIM 4
#define B_DIM 64
#define N_DIM 256
#define D_DIM 512
#define E_DIM 512
#define M_DIM (T_DIM * B_DIM * N_DIM)   // 65536 rows
#define BN_EPS 1e-5f

// ---------------------------------------------------------------------------
// TDM: 1-D contiguous copy global -> LDS, n_dwords 4-byte elements.
// D# per CDNA5 ISA.  This toolchain exposes the 6-arg
// builtin (4 descriptor groups + extra group + cpol).
//   g0: count=1 | lds_addr | global_addr | type=2
//   g1: data_size=4B, tensor_dim0 = tile_dim0 = n_dwords, tensor_dim1=tile_dim1=1
// ---------------------------------------------------------------------------
__device__ __forceinline__ void tdm_load_1d(const void* gsrc, void* lds_dst,
                                            unsigned int n_dwords) {
    unsigned long long ga = (unsigned long long)(uintptr_t)gsrc;
    unsigned int       la = (unsigned int)(uintptr_t)lds_dst;   // addr[31:0] = LDS offset
    v4u g0;
    g0[0] = 1u;                                            // count=1, user descriptor
    g0[1] = la;                                            // lds_addr
    g0[2] = (unsigned int)(ga & 0xFFFFFFFFu);              // global_addr[31:0]
    g0[3] = (unsigned int)((ga >> 32) & 0x01FFFFFFu)       // global_addr[56:32]
          | (2u << 30);                                    // type=2 ("image")
    v8i g1;
    g1[0] = (int)(2u << 16);                               // data_size=2 -> 4 bytes
    g1[1] = (int)((n_dwords & 0xFFFFu) << 16);             // tensor_dim0[15:0] @ bits63:48
    g1[2] = (int)((n_dwords >> 16) & 0xFFFFu)              // tensor_dim0[31:16]
          | (1 << 16);                                     // tensor_dim1=1 @ bits95:80
    g1[3] = (int)((n_dwords & 0xFFFFu) << 16);             // tile_dim0 @ bits127:112
    g1[4] = 1;                                             // tile_dim1=1
    g1[5] = (int)n_dwords;                                 // tensor_dim0_stride[31:0]
    g1[6] = 0;                                             // stride hi, dim1_stride lo
    g1[7] = 0;
    v4i g2 = {0, 0, 0, 0};
    v4i g3 = {0, 0, 0, 0};
    v8i g4 = {0, 0, 0, 0, 0, 0, 0, 0};
    __builtin_amdgcn_tensor_load_to_lds(g0, g1, g2, g3, g4, 0);
}

// ---------------------------------------------------------------------------
// Kernel 1: zero BN stat accumulators (ws is poisoned by the harness)
// ---------------------------------------------------------------------------
__global__ void zero_stats_kernel(float* __restrict__ stats) {
    stats[threadIdx.x] = 0.0f;   // sum[512] + sumsq[512]
}

// ---------------------------------------------------------------------------
// Kernel 2: transpose mx [N,B,T,D] -> m-major [M,D] and split fp32 into
// hi/lo f16 pair (done ONCE per element instead of 32x inside the GEMM).
// 2 elements per thread: float2 in, packed 2xf16 out.
// ---------------------------------------------------------------------------
__global__ __launch_bounds__(256) void convert_mx_kernel(
    const float* __restrict__ mx,
    _Float16* __restrict__ ahi, _Float16* __restrict__ alo)
{
    const size_t p = ((size_t)blockIdx.x * 256 + threadIdx.x) * 2;
    const int m = (int)(p >> 9);
    const int d = (int)(p & 511);
    const int t = m / (B_DIM * N_DIM);
    const int b = (m / N_DIM) % B_DIM;
    const int n = m % N_DIM;
    const float2 v = *(const float2*)&mx[(((size_t)n * B_DIM + b) * T_DIM + t) * D_DIM + d];
    _Float16 h0 = (_Float16)v.x, h1 = (_Float16)v.y;
    union { _Float16 h[2]; unsigned int u; } hi, lo;
    hi.h[0] = h0;                      hi.h[1] = h1;
    lo.h[0] = (_Float16)(v.x - (float)h0);
    lo.h[1] = (_Float16)(v.y - (float)h1);
    *(unsigned int*)&ahi[p] = hi.u;
    *(unsigned int*)&alo[p] = lo.u;
}

// ---------------------------------------------------------------------------
// Kernel 3: split W into hi/lo f16 (same layout, [E,D] row-major)
// ---------------------------------------------------------------------------
__global__ __launch_bounds__(256) void convert_w_kernel(
    const float* __restrict__ Wm,
    _Float16* __restrict__ whi, _Float16* __restrict__ wlo)
{
    const size_t p = ((size_t)blockIdx.x * 256 + threadIdx.x) * 2;
    const float2 v = *(const float2*)&Wm[p];
    _Float16 h0 = (_Float16)v.x, h1 = (_Float16)v.y;
    union { _Float16 h[2]; unsigned int u; } hi, lo;
    hi.h[0] = h0;                      hi.h[1] = h1;
    lo.h[0] = (_Float16)(v.x - (float)h0);
    lo.h[1] = (_Float16)(v.y - (float)h1);
    *(unsigned int*)&whi[p] = hi.u;
    *(unsigned int*)&wlo[p] = lo.u;
}

// ---------------------------------------------------------------------------
// Kernel 4: GEMM  h[m,e] = sum_d A[m,d] * W[e,d]   (A pre-transposed, f16x3)
//   Block = 8 waves; wave w owns 16x16 tile (m0, e0 = 128*by + 16*w).
//   A tile (16 rows, hi+lo = 32KB, contiguous) staged to LDS by the TDM;
//   B fragments read straight from the f16 W buffers (L2-resident).
//   3 WMMAs per 32-wide K step: hi*hi + hi*lo + lo*hi ~ fp32 accuracy.
// ---------------------------------------------------------------------------
__global__ __launch_bounds__(256) void gemm_tdm_wmma_kernel(
    const _Float16* __restrict__ ahi, const _Float16* __restrict__ alo,
    const _Float16* __restrict__ whi, const _Float16* __restrict__ wlo,
    float* __restrict__ h)
{
    __shared__ _Float16 sAhi[16 * D_DIM];   // 16 KB
    __shared__ _Float16 sAlo[16 * D_DIM];   // 16 KB

    const int lane = threadIdx.x & 31;
    const int wave = threadIdx.x >> 5;
    const int l    = lane & 15;
    const int hlf  = lane >> 4;
    const int m0   = blockIdx.x * 16;

    // Wave 0 issues both TDM tile loads and drains TENSORcnt; the workgroup
    // barrier then publishes the LDS tile to the other 7 waves.
    if (threadIdx.x < 32) {
        tdm_load_1d(ahi + (size_t)m0 * D_DIM, sAhi, (16 * D_DIM * 2) / 4);
        tdm_load_1d(alo + (size_t)m0 * D_DIM, sAlo, (16 * D_DIM * 2) / 4);
        __builtin_amdgcn_s_wait_tensorcnt(0);
    }
    __syncthreads();

    const int e = blockIdx.y * 128 + wave * 16 + l;
    const _Float16* __restrict__ bhrow = whi + (size_t)e * D_DIM;
    const _Float16* __restrict__ blrow = wlo + (size_t)e * D_DIM;

    v8f c = {};

    for (int kk = 0; kk < D_DIM; kk += 32) {
        // A (16-bit, 16x32): lanes 0-15 carry K = d0..d0+7, d0+16..d0+23 (d0=kk);
        // lanes 16-31 the same shifted by 8.
        const int d0  = kk + hlf * 8;
        // B (16-bit, 32x16): same K order as A (kk..kk+7, kk+16..kk+23;
        // lanes 16-31 shifted by 8).
        const int d0b = kk + hlf * 8;

        const v8h ah0 = *(const v8h*)&sAhi[l * D_DIM + d0];
        const v8h ah1 = *(const v8h*)&sAhi[l * D_DIM + d0 + 16];
        const v8h al0 = *(const v8h*)&sAlo[l * D_DIM + d0];
        const v8h al1 = *(const v8h*)&sAlo[l * D_DIM + d0 + 16];
        const v8h bh0 = *(const v8h*)&bhrow[d0b];
        const v8h bh1 = *(const v8h*)&bhrow[d0b + 16];
        const v8h bl0 = *(const v8h*)&blrow[d0b];
        const v8h bl1 = *(const v8h*)&blrow[d0b + 16];

        v16h AH, AL, BH, BL;
        #pragma unroll
        for (int i = 0; i < 8; ++i) {
            AH[i] = ah0[i]; AH[8 + i] = ah1[i];
            AL[i] = al0[i]; AL[8 + i] = al1[i];
            BH[i] = bh0[i]; BH[8 + i] = bh1[i];
            BL[i] = bl0[i]; BL[8 + i] = bl1[i];
        }

        c = __builtin_amdgcn_wmma_f32_16x16x32_f16(false, AH, false, BH,
                                                   (short)0, c, false, false);
        c = __builtin_amdgcn_wmma_f32_16x16x32_f16(false, AH, false, BL,
                                                   (short)0, c, false, false);
        c = __builtin_amdgcn_wmma_f32_16x16x32_f16(false, AL, false, BH,
                                                   (short)0, c, false, false);
    }

    // C/D layout: VGPR i holds M = i + 8*hlf, N = lane&15
    #pragma unroll
    for (int i = 0; i < 8; ++i) {
        const int row = m0 + hlf * 8 + i;
        h[(size_t)row * E_DIM + e] = c[i];
    }
}

// ---------------------------------------------------------------------------
// Kernel 5: BN stats — per-channel sum / sumsq over all M rows.
// ---------------------------------------------------------------------------
__global__ __launch_bounds__(512) void bn_stats_kernel(
    const float* __restrict__ h, float* __restrict__ sum,
    float* __restrict__ sumsq)
{
    const int e  = threadIdx.x;
    const int r0 = blockIdx.x * (M_DIM / 256);
    float s = 0.0f, s2 = 0.0f;
    for (int r = 0; r < M_DIM / 256; ++r) {
        const float v = h[(size_t)(r0 + r) * E_DIM + e];
        s  += v;
        s2 += v * v;
    }
    atomicAdd(&sum[e], s);
    atomicAdd(&sumsq[e], s2);
}

// ---------------------------------------------------------------------------
// Kernel 6: fold mean/var/gamma/beta into per-channel scale & shift
// ---------------------------------------------------------------------------
__global__ void bn_finalize_kernel(
    const float* __restrict__ sum, const float* __restrict__ sumsq,
    const float* __restrict__ gamma, const float* __restrict__ beta,
    float* __restrict__ scale, float* __restrict__ shift)
{
    const int e = threadIdx.x;
    const float inv_m = 1.0f / (float)M_DIM;
    const float mean  = sum[e] * inv_m;
    const float var   = sumsq[e] * inv_m - mean * mean;   // biased var
    const float sc    = gamma[e] * rsqrtf(var + BN_EPS);
    scale[e] = sc;
    shift[e] = beta[e] - mean * sc;
}

// ---------------------------------------------------------------------------
// Kernel 7: LIF recurrence over T + output mask  out = x * spike
// ---------------------------------------------------------------------------
__global__ __launch_bounds__(256) void lif_out_kernel(
    const float* __restrict__ h, const float* __restrict__ x,
    const float* __restrict__ scale, const float* __restrict__ shift,
    float* __restrict__ out)
{
    const size_t p = (size_t)blockIdx.x * 256 + threadIdx.x;
    const int e = (int)(p & (E_DIM - 1));
    const float sc = scale[e];
    const float sh = shift[e];
    const size_t S = (size_t)B_DIM * N_DIM * E_DIM;

    float v = 0.0f;
    #pragma unroll
    for (int t = 0; t < T_DIM; ++t) {
        const float hn = h[(size_t)t * S + p] * sc + sh;     // BN
        v = 0.5f * (v + hn);                                  // v += (hn-v)/tau
        const float sp = (v >= 0.5f) ? 1.0f : 0.0f;           // Heaviside
        out[(size_t)t * S + p] = x[(size_t)t * S + p] * sp;   // x * spike
        v *= (1.0f - sp);                                     // hard reset
    }
}

// ---------------------------------------------------------------------------
extern "C" void kernel_launch(void* const* d_in, const int* in_sizes, int n_in,
                              void* d_out, int out_size, void* d_ws, size_t ws_size,
                              hipStream_t stream) {
    const float* x     = (const float*)d_in[0];   // [T,B,N,D]
    const float* mx    = (const float*)d_in[1];   // [N,B,T,D]
    const float* Wm    = (const float*)d_in[2];   // [D,D]
    const float* gamma = (const float*)d_in[3];   // [D]
    const float* beta  = (const float*)d_in[4];   // [D]
    float*       out   = (float*)d_out;           // [T,B,N,D]

    // Workspace layout (bytes):
    //   h     : M*E*4        = 128 MiB
    //   ahi   : M*D*2        =  64 MiB
    //   alo   : M*D*2        =  64 MiB
    //   whi   : E*D*2        = 512 KiB
    //   wlo   : E*D*2        = 512 KiB
    //   stats : 2048 floats
    char* ws = (char*)d_ws;
    float*    h     = (float*)ws;                       ws += (size_t)M_DIM * E_DIM * 4;
    _Float16* ahi   = (_Float16*)ws;                    ws += (size_t)M_DIM * D_DIM * 2;
    _Float16* alo   = (_Float16*)ws;                    ws += (size_t)M_DIM * D_DIM * 2;
    _Float16* whi   = (_Float16*)ws;                    ws += (size_t)E_DIM * D_DIM * 2;
    _Float16* wlo   = (_Float16*)ws;                    ws += (size_t)E_DIM * D_DIM * 2;
    float*    stats = (float*)ws;
    float*    sum   = stats;
    float*    sumsq = stats + 512;
    float*    scale = stats + 1024;
    float*    shift = stats + 1536;

    hipLaunchKernelGGL(zero_stats_kernel, dim3(1), dim3(1024), 0, stream, stats);

    hipLaunchKernelGGL(convert_mx_kernel, dim3((M_DIM * (D_DIM / 2)) / 256),
                       dim3(256), 0, stream, mx, ahi, alo);
    hipLaunchKernelGGL(convert_w_kernel, dim3((E_DIM * (D_DIM / 2)) / 256),
                       dim3(256), 0, stream, Wm, whi, wlo);

    hipLaunchKernelGGL(gemm_tdm_wmma_kernel,
                       dim3(M_DIM / 16, E_DIM / 128), dim3(256), 0, stream,
                       ahi, alo, whi, wlo, h);

    hipLaunchKernelGGL(bn_stats_kernel, dim3(256), dim3(512), 0, stream,
                       h, sum, sumsq);

    hipLaunchKernelGGL(bn_finalize_kernel, dim3(1), dim3(512), 0, stream,
                       sum, sumsq, gamma, beta, scale, shift);

    hipLaunchKernelGGL(lif_out_kernel,
                       dim3((B_DIM * N_DIM * E_DIM) / 256), dim3(256), 0, stream,
                       h, x, scale, shift, out);
}
